// GatedDeltaNet_25640954757703
// MI455X (gfx1250) — compile-verified, matrix operand order checked by reading, not run on hardware
//
#include <hip/hip_runtime.h>
#include <hip/hip_bf16.h>
#include <math.h>

#define T_SEQ 2048
#define HID   2048
#define H     32
#define DH    64
#define KDIM  2048
#define CONVK 4
#define EPSF  1e-6f

typedef __bf16 bf16;
typedef bf16  v8bf  __attribute__((ext_vector_type(8)));
typedef bf16  v16bf __attribute__((ext_vector_type(16)));
typedef float v8f   __attribute__((ext_vector_type(8)));

__device__ __forceinline__ float sigmoidf_(float x) { return 1.0f / (1.0f + __expf(-x)); }
__device__ __forceinline__ float siluf_(float x)    { return x * sigmoidf_(x); }

// ---------------------------------------------------------------- cast f32 -> bf16
__global__ void __launch_bounds__(256)
cast_f32_bf16(const float* __restrict__ in, bf16* __restrict__ out, int n) {
  int i = blockIdx.x * blockDim.x + threadIdx.x;
  const int stride = gridDim.x * blockDim.x;
  for (; i < n; i += stride) out[i] = (bf16)in[i];
}

// ---------------------------------------------------------------- WMMA bf16 GEMM
// C[M,N] (f32, row-major) = A[M,K] (bf16, row-major) * B[N,K]^T (bf16, row-major)
// CDNA5 16-bit A/B fragment layout: lane<16 holds K {0..7,16..23}, lane>=16 holds
// K {8..15,24..31}; each contiguous K-octet is one 16B load.
__device__ __forceinline__ v16bf load_frag32(const bf16* __restrict__ p, int hi) {
  const v8bf lo = *(const v8bf*)(p + (hi ? 8 : 0));
  const v8bf hh = *(const v8bf*)(p + (hi ? 24 : 16));
  return __builtin_shufflevector(lo, hh, 0, 1, 2, 3, 4, 5, 6, 7,
                                 8, 9, 10, 11, 12, 13, 14, 15);
}

__global__ void __launch_bounds__(256)
gemm_bf16_wmma(const bf16* __restrict__ A, const bf16* __restrict__ B,
               float* __restrict__ C, int M, int N, int K) {
  const int lane  = threadIdx.x & 31;
  const int wid   = threadIdx.x >> 5;
  const int l16   = lane & 15;
  const int hi    = lane >> 4;
  const int waveN = wid & 1;       // 2 waves across N
  const int waveM = wid >> 1;      // 4 waves across M
  const int m0 = blockIdx.y * 256 + waveM * 64;
  const int n0 = blockIdx.x * 128 + waveN * 64;

  const bf16* Ar[4];
  const bf16* Br[4];
#pragma unroll
  for (int i = 0; i < 4; ++i) {
    Ar[i] = A + (size_t)(m0 + i * 16 + l16) * K;
    Br[i] = B + (size_t)(n0 + i * 16 + l16) * K;
  }

  v8f acc[4][4] = {};

  for (int k0 = 0; k0 < K; k0 += 32) {
    v16bf af[4], bfr[4];
#pragma unroll
    for (int i = 0; i < 4; ++i) {
      af[i]  = load_frag32(Ar[i] + k0, hi);
      bfr[i] = load_frag32(Br[i] + k0, hi);
    }
    // hint the K data 512 elements (16 steps) ahead into cache (global_prefetch_b8)
    __builtin_prefetch(Ar[0] + k0 + 512, 0, 3);
    __builtin_prefetch(Br[0] + k0 + 512, 0, 3);
#pragma unroll
    for (int mi = 0; mi < 4; ++mi)
#pragma unroll
      for (int ni = 0; ni < 4; ++ni)
        acc[mi][ni] = __builtin_amdgcn_wmma_f32_16x16x32_bf16(
            false, af[mi], false, bfr[ni], (short)0, acc[mi][ni], false, false);
  }

#pragma unroll
  for (int mi = 0; mi < 4; ++mi)
#pragma unroll
    for (int ni = 0; ni < 4; ++ni) {
      const int n = n0 + ni * 16 + l16;
#pragma unroll
      for (int r = 0; r < 8; ++r) {
        const int m = m0 + mi * 16 + hi * 8 + r;   // D layout: VGPR r -> M = r + 8*hi
        C[(size_t)m * N + n] = acc[mi][ni][r];
      }
    }
}

// ---------------------------------------------------------------- beta / alpha
__global__ void __launch_bounds__(256)
beta_alpha_kernel(const float* __restrict__ x,
                  const float* __restrict__ Wb,  const float* __restrict__ bb,
                  const float* __restrict__ Wgk, const float* __restrict__ bgk,
                  const float* __restrict__ A_log, const float* __restrict__ dt_bias,
                  float* __restrict__ beta, float* __restrict__ alpha) {
  const int t = blockIdx.x * (blockDim.x >> 5) + (threadIdx.x >> 5);
  const int n = threadIdx.x & 31;   // head index (H==32 == wave32)
  if (t >= T_SEQ) return;
  const float* xr = x   + (size_t)t * HID;
  const float* wb = Wb  + (size_t)n * HID;
  const float* wg = Wgk + (size_t)n * HID;
  float d1 = 0.0f, d2 = 0.0f;
  for (int c = 0; c < HID; c += 4) {
    const float4 xv = *(const float4*)(xr + c);
    const float4 w1 = *(const float4*)(wb + c);
    const float4 w2 = *(const float4*)(wg + c);
    d1 += xv.x * w1.x + xv.y * w1.y + xv.z * w1.z + xv.w * w1.w;
    d2 += xv.x * w2.x + xv.y * w2.y + xv.z * w2.z + xv.w * w2.w;
  }
  beta[t * H + n] = sigmoidf_(d1 + bb[n]);
  const float z  = d2 + bgk[n] + dt_bias[n];
  const float sp = (z > 20.0f) ? z : log1pf(__expf(z));
  alpha[t * H + n] = __expf(-__expf(A_log[n]) * sp);
}

// ---------------------------------------------------------------- conv4+SiLU (+RoPE+l2norm)
// one wave per (t, h); lane owns channel pair (2*lane, 2*lane+1) == RoPE pair
__global__ void __launch_bounds__(256)
conv_rope_kernel(const float* __restrict__ pre,
                 const float* __restrict__ wconv, const float* __restrict__ bconv,
                 float* __restrict__ out, int doRope) {
  const int gw = blockIdx.x * (blockDim.x >> 5) + (threadIdx.x >> 5);
  if (gw >= T_SEQ * H) return;
  const int t = gw / H;
  const int h = gw % H;
  const int lane = threadIdx.x & 31;
  const int c0 = h * DH + 2 * lane;

  float y0 = bconv[c0], y1 = bconv[c0 + 1];
  const float* w0 = wconv + (size_t)c0 * CONVK;
  const float* w1 = w0 + CONVK;
#pragma unroll
  for (int j = 0; j < CONVK; ++j) {
    const int ts = t - (CONVK - 1) + j;
    if (ts >= 0) {
      const float* pr = pre + (size_t)ts * KDIM + c0;
      y0 += w0[j] * pr[0];
      y1 += w1[j] * pr[1];
    }
  }
  y0 = siluf_(y0);
  y1 = siluf_(y1);

  float* dst = out + (size_t)t * KDIM + h * DH;
  if (!doRope) {                       // uniform branch (v path)
    dst[2 * lane]     = y0;
    dst[2 * lane + 1] = y1;
    return;
  }
  // YaRN rope: angle index j = (2*lane) & 31
  const int   j       = (2 * lane) & 31;
  const float inv     = __powf(10000.0f, -(float)j * (1.0f / 32.0f));
  const float wavelen = 6.283185307179586f / inv;
  float ramp = (wavelen - 1.0f) * (1.0f / 31.0f);
  ramp = fminf(fmaxf(ramp, 0.0f), 1.0f);
  const float scale = 1.0f + 31.0f * ramp;
  const float ang = (float)t * inv / scale;
  float ss, cc;
  __sincosf(ang, &ss, &cc);
  const float o0 = y0 * cc - y1 * ss;   // -> index lane
  const float o1 = y0 * ss + y1 * cc;   // -> index lane+32
  float sq = o0 * o0 + o1 * o1;
#pragma unroll
  for (int m = 16; m >= 1; m >>= 1) sq += __shfl_xor(sq, m, 32);
  const float r = rsqrtf(sq + EPSF);
  dst[lane]      = o0 * r;
  dst[lane + 32] = o1 * r;
}

// ---------------------------------------------------------------- delta-rule scan
// one wave per head; 64x64 state in registers: lane owns columns {lane, lane+32}
__global__ void __launch_bounds__(32)
scan_kernel(const float* __restrict__ q, const float* __restrict__ k,
            const float* __restrict__ v, const float* __restrict__ beta,
            const float* __restrict__ alpha, float* __restrict__ o) {
  const int h = blockIdx.x;
  const int lane = threadIdx.x & 31;
  float s0[64], s1[64];
#pragma unroll
  for (int i = 0; i < 64; ++i) { s0[i] = 0.0f; s1[i] = 0.0f; }

  for (int t = 0; t < T_SEQ; ++t) {
    const size_t base = (size_t)t * KDIM + h * DH;
    const float q0 = q[base + lane], q1 = q[base + lane + 32];
    const float k0 = k[base + lane], k1 = k[base + lane + 32];
    const float v0 = v[base + lane], v1 = v[base + lane + 32];
    const float a = alpha[t * H + h];
    const float b = beta[t * H + h];

    float e0 = v0, e1 = v1;                 // err = v - k . (a*S)
#pragma unroll
    for (int kk = 0; kk < 64; ++kk) {
      const float kb = __shfl(kk < 32 ? k0 : k1, kk & 31, 32);
      s0[kk] *= a; s1[kk] *= a;
      e0 -= kb * s0[kk];
      e1 -= kb * s1[kk];
    }
    e0 *= b; e1 *= b;

    float o0 = 0.0f, o1 = 0.0f;             // S += k (x) err ; o = q . S
#pragma unroll
    for (int kk = 0; kk < 64; ++kk) {
      const float kb = __shfl(kk < 32 ? k0 : k1, kk & 31, 32);
      const float qb = __shfl(kk < 32 ? q0 : q1, kk & 31, 32);
      s0[kk] += kb * e0; s1[kk] += kb * e1;
      o0 += qb * s0[kk];
      o1 += qb * s1[kk];
    }
    o[base + lane]      = o0;
    o[base + lane + 32] = o1;
  }
}

// ---------------------------------------------------------------- gated RMS-norm -> bf16
__global__ void __launch_bounds__(256)
out_norm_kernel(const float* __restrict__ o_scan, const float* __restrict__ v,
                const float* __restrict__ gpre, const float* __restrict__ D,
                const float* __restrict__ onorm_w, bf16* __restrict__ o_bf) {
  const int gw = blockIdx.x * (blockDim.x >> 5) + (threadIdx.x >> 5);
  if (gw >= T_SEQ * H) return;
  const int t = gw / H, h = gw % H;
  const int lane = threadIdx.x & 31;
  const size_t base = (size_t)t * KDIM + h * DH;
  const float dd = D[h];
  const float ov0 = o_scan[base + lane]      + dd * v[base + lane];
  const float ov1 = o_scan[base + lane + 32] + dd * v[base + lane + 32];
  float sq = ov0 * ov0 + ov1 * ov1;
#pragma unroll
  for (int m = 16; m >= 1; m >>= 1) sq += __shfl_xor(sq, m, 32);
  const float r  = rsqrtf(sq * (1.0f / (float)DH) + EPSF);
  const float z0 = onorm_w[lane]      * ov0 * r;
  const float z1 = onorm_w[lane + 32] * ov1 * r;
  o_bf[base + lane]      = (bf16)(gpre[base + lane]      * siluf_(z0));
  o_bf[base + lane + 32] = (bf16)(gpre[base + lane + 32] * siluf_(z1));
}

// ---------------------------------------------------------------- launch
extern "C" void kernel_launch(void* const* d_in, const int* in_sizes, int n_in,
                              void* d_out, int out_size, void* d_ws, size_t ws_size,
                              hipStream_t stream) {
  (void)in_sizes; (void)n_in; (void)out_size; (void)ws_size;
  const float* x       = (const float*)d_in[0];
  const float* Wq      = (const float*)d_in[1];
  const float* Wk      = (const float*)d_in[2];
  const float* Wv      = (const float*)d_in[3];
  const float* Wg      = (const float*)d_in[4];
  const float* Wo      = (const float*)d_in[5];
  const float* Wb      = (const float*)d_in[6];
  const float* bb      = (const float*)d_in[7];
  const float* Wgk     = (const float*)d_in[8];
  const float* bgk     = (const float*)d_in[9];
  const float* wq_conv = (const float*)d_in[10];
  const float* bq_conv = (const float*)d_in[11];
  const float* wk_conv = (const float*)d_in[12];
  const float* bk_conv = (const float*)d_in[13];
  const float* wv_conv = (const float*)d_in[14];
  const float* bv_conv = (const float*)d_in[15];
  const float* A_log   = (const float*)d_in[16];
  const float* D       = (const float*)d_in[17];
  const float* dt_bias = (const float*)d_in[18];
  const float* onorm_w = (const float*)d_in[19];
  float* out = (float*)d_out;

  const int NE = T_SEQ * KDIM;             // 2048*2048
  size_t off = 0;
  char* ws = (char*)d_ws;
  auto carve = [&](size_t bytes) -> void* {
    void* p = ws + off;
    off += (bytes + 255) & ~(size_t)255;
    return p;
  };
  bf16* x_bf   = (bf16*)carve((size_t)NE * 2);
  bf16* Wq_bf  = (bf16*)carve((size_t)NE * 2);
  bf16* Wk_bf  = (bf16*)carve((size_t)NE * 2);
  bf16* Wv_bf  = (bf16*)carve((size_t)NE * 2);
  bf16* Wg_bf  = (bf16*)carve((size_t)NE * 2);
  bf16* Wo_bf  = (bf16*)carve((size_t)NE * 2);
  float* pre_q = (float*)carve((size_t)NE * 4);
  float* pre_k = (float*)carve((size_t)NE * 4);
  float* pre_v = (float*)carve((size_t)NE * 4);
  float* gpre  = (float*)carve((size_t)NE * 4);
  float* qf    = (float*)carve((size_t)NE * 4);
  float* kf    = (float*)carve((size_t)NE * 4);
  float* vf    = (float*)carve((size_t)NE * 4);
  float* betab = (float*)carve((size_t)T_SEQ * H * 4);
  float* alphab= (float*)carve((size_t)T_SEQ * H * 4);
  float* o_scan = pre_q;                   // pre_q dead after conv stage
  bf16*  o_bf   = (bf16*)pre_k;            // pre_k dead after conv stage

  // 1) casts to bf16
  cast_f32_bf16<<<1024, 256, 0, stream>>>(x,  x_bf,  NE);
  cast_f32_bf16<<<1024, 256, 0, stream>>>(Wq, Wq_bf, NE);
  cast_f32_bf16<<<1024, 256, 0, stream>>>(Wk, Wk_bf, NE);
  cast_f32_bf16<<<1024, 256, 0, stream>>>(Wv, Wv_bf, NE);
  cast_f32_bf16<<<1024, 256, 0, stream>>>(Wg, Wg_bf, NE);
  cast_f32_bf16<<<1024, 256, 0, stream>>>(Wo, Wo_bf, NE);

  // 2) projection GEMMs (WMMA):  pre = x @ W^T
  const dim3 ggrid(KDIM / 128, T_SEQ / 256);
  gemm_bf16_wmma<<<ggrid, 256, 0, stream>>>(x_bf, Wq_bf, pre_q, T_SEQ, KDIM, HID);
  gemm_bf16_wmma<<<ggrid, 256, 0, stream>>>(x_bf, Wk_bf, pre_k, T_SEQ, KDIM, HID);
  gemm_bf16_wmma<<<ggrid, 256, 0, stream>>>(x_bf, Wv_bf, pre_v, T_SEQ, KDIM, HID);
  gemm_bf16_wmma<<<ggrid, 256, 0, stream>>>(x_bf, Wg_bf, gpre,  T_SEQ, KDIM, HID);

  // 3) beta / alpha (small N=32 projections, VALU)
  beta_alpha_kernel<<<T_SEQ / 8, 256, 0, stream>>>(x, Wb, bb, Wgk, bgk, A_log,
                                                   dt_bias, betab, alphab);

  // 4) conv + SiLU (+ RoPE + l2norm for q, k)
  const int nWaves = T_SEQ * H;            // 65536 waves, 8 per block
  conv_rope_kernel<<<nWaves / 8, 256, 0, stream>>>(pre_q, wq_conv, bq_conv, qf, 1);
  conv_rope_kernel<<<nWaves / 8, 256, 0, stream>>>(pre_k, wk_conv, bk_conv, kf, 1);
  conv_rope_kernel<<<nWaves / 8, 256, 0, stream>>>(pre_v, wv_conv, bv_conv, vf, 0);

  // 5) sequential delta-rule scan, one wave per head, state in registers
  scan_kernel<<<H, 32, 0, stream>>>(qf, kf, vf, betab, alphab, o_scan);

  // 6) gated RMS-norm -> bf16
  out_norm_kernel<<<nWaves / 8, 256, 0, stream>>>(o_scan, vf, gpre, D, onorm_w, o_bf);

  // 7) output GEMM (WMMA): out = o @ Wo^T
  gemm_bf16_wmma<<<ggrid, 256, 0, stream>>>(o_bf, Wo_bf, out, T_SEQ, HID, KDIM);
}
